// LiTENBlock_46531675685328
// MI455X (gfx1250) — matrix operand order repaired and checked by reading, not compiled, not run on hardware
//
#include <hip/hip_runtime.h>
#include <hip/hip_bf16.h>

// ---------------------------------------------------------------------------
// LiTEN-style equivariant GNN forward for MI455X (gfx1250, wave32).
// GEMMs run through V_WMMA_F32_16X16X32_BF16 (bf16 operands, f32 accum):
// at 23.3 TB/s the forward is bandwidth-bound (~3 GB traffic), so we pick the
// densest matrix path and fuse bias/SiLU epilogues. The K=256 panel is staged
// in LDS once per block (50 KB of the 320 KB WGP LDS), so the 8 WMMAs issue
// back-to-back with no intervening barriers. Per-head attention reductions
// map 1:1 onto wave32 (HD == 32). Segment sums use f32 global atomics.
// ---------------------------------------------------------------------------

#define Cc   256
#define NRBF 32
#define CUTR 5.0f
#define LDK  264   // padded K stride (bf16 elems): keeps b128 frags aligned,
                   // row stride 132 dwords == 4 mod 64 banks -> conflict-free

typedef __attribute__((ext_vector_type(16))) __bf16 v16bf;
typedef __attribute__((ext_vector_type(8)))  float  v8f;

__device__ __forceinline__ unsigned short f2bf(float f) {
  unsigned int u = __float_as_uint(f);
  unsigned int r = u + 0x7FFFu + ((u >> 16) & 1u);  // round-to-nearest-even
  return (unsigned short)(r >> 16);
}
// branch-free SiLU: x * v_rcp_f32(1+exp(-x))
__device__ __forceinline__ float silu_f(float x) {
  return x * __builtin_amdgcn_rcpf(1.f + __expf(-x));
}

// ---------------------------------------------------------------------------
// GEMM: out[M][NO] = act(in[M][256] @ W[256][NO] + bias), act: 0=none 1=silu
// block = 256 threads (8 waves); block tile 32(M) x 64(N); full K panel in
// LDS as bf16; each wave owns one 16x16 tile -> 8 unrolled WMMAs.
// ---------------------------------------------------------------------------
__global__ __launch_bounds__(256)
void k_gemm(const float* __restrict__ in, const float* __restrict__ W,
            const float* __restrict__ bias, float* __restrict__ out,
            int M, int NO, int act) {
  __shared__ unsigned short As[32][LDK];  // [m][k]
  __shared__ unsigned short Bs[64][LDK];  // [n][k] (transposed W panel)

  const int tid  = threadIdx.x;
  const int m0   = blockIdx.x * 32;
  const int n0   = blockIdx.y * 64;
  const int lane = tid & 31;
  const int wv   = tid >> 5;
  const int mbase = (wv >> 2) * 16;  // 0 or 16
  const int nbase = (wv & 3) * 16;   // 0..48

  {  // stage A: 32 rows x 256 k; 8 threads/row, 8 float4 each
    int r  = tid >> 3;
    int gm = m0 + r;
    bool ok = gm < M;
    const float4* src = (const float4*)(in + (size_t)gm * Cc);
#pragma unroll
    for (int j = 0; j < 8; ++j) {
      int cj = (tid & 7) + 8 * j;
      float4 f = ok ? src[cj] : make_float4(0.f, 0.f, 0.f, 0.f);
      int k0 = cj << 2;
      As[r][k0 + 0] = f2bf(f.x); As[r][k0 + 1] = f2bf(f.y);
      As[r][k0 + 2] = f2bf(f.z); As[r][k0 + 3] = f2bf(f.w);
    }
  }
  {  // stage B transposed: 256 k x 64 n; thread -> 8 k-rows, 8 n each
    int nc = (tid & 7) << 3;
#pragma unroll
    for (int j = 0; j < 8; ++j) {
      int k = (tid >> 3) + 32 * j;
      const float4* wp = (const float4*)(W + (size_t)k * NO + n0 + nc);
      float4 f0 = wp[0], f1 = wp[1];
      Bs[nc + 0][k] = f2bf(f0.x); Bs[nc + 1][k] = f2bf(f0.y);
      Bs[nc + 2][k] = f2bf(f0.z); Bs[nc + 3][k] = f2bf(f0.w);
      Bs[nc + 4][k] = f2bf(f1.x); Bs[nc + 5][k] = f2bf(f1.y);
      Bs[nc + 6][k] = f2bf(f1.z); Bs[nc + 7][k] = f2bf(f1.w);
    }
  }
  __syncthreads();

  // Fragment layouts (ISA 7.12.2):
  //  A 16x32: lanes 0-15 K={0..7,16..23}, lanes 16-31 K={8..15,24..31}, M=lane&15
  //  B 32x16: lanes 0-15 K={0..7,16..23}, lanes 16-31 K={8..15,24..31}, N=lane&15
  const int am  = mbase + (lane & 15);
  const int akb = (lane < 16) ? 0 : 8;
  const int bn  = nbase + (lane & 15);
  const int bkb = (lane < 16) ? 0 : 8;

  v8f acc = {};
#pragma unroll
  for (int kk = 0; kk < Cc; kk += 32) {
    union { v16bf v; uint4 q[2]; } au, bu;
    au.q[0] = *(const uint4*)&As[am][kk + akb];
    au.q[1] = *(const uint4*)&As[am][kk + 16 + akb];
    bu.q[0] = *(const uint4*)&Bs[bn][kk + bkb];
    bu.q[1] = *(const uint4*)&Bs[bn][kk + 16 + bkb];
    acc = __builtin_amdgcn_wmma_f32_16x16x32_bf16(false, au.v, false, bu.v,
                                                  (short)0, acc, false, false);
  }

  // C/D layout: VGPR i, lanes 0-15 -> M=i, lanes 16-31 -> M=8+i; N=lane&15.
  int col   = n0 + nbase + (lane & 15);
  int rbase = m0 + mbase + ((lane < 16) ? 0 : 8);
  float bv = bias ? bias[col] : 0.f;
#pragma unroll
  for (int i = 0; i < 8; ++i) {
    int r = rbase + i;
    if (r < M) {
      float v = acc[i] + bv;
      if (act) v = silu_f(v);
      out[(size_t)r * NO + col] = v;
    }
  }
}

// --------------------------- elementwise / graph ---------------------------

__global__ void k_zero(float* p, size_t n) {
  size_t i = (size_t)blockIdx.x * blockDim.x + threadIdx.x;
  size_t stride = (size_t)gridDim.x * blockDim.x;
  for (; i < n; i += stride) p[i] = 0.f;
}

__global__ void k_embed(const int* __restrict__ sp, const float* __restrict__ tbl,
                        float* __restrict__ ns, int N) {
  int i = blockIdx.x * blockDim.x + threadIdx.x;
  if (i < N * Cc) {
    int n = i / Cc, c = i % Cc;
    ns[i] = tbl[sp[n] * Cc + c];
  }
}

// RBF expansion + projection onto C channels.  one block per edge.
__global__ void k_rbf_ef(const float* __restrict__ dist, const float* __restrict__ eeW,
                         const float* __restrict__ eeb, float* __restrict__ ef,
                         float* __restrict__ cut, int E) {
  __shared__ float rb[NRBF];
  int e = blockIdx.x, tid = threadIdx.x;
  float d  = dist[e];
  float cc = (d < CUTR) ? 0.5f * (__cosf(3.14159265358979f * d / CUTR) + 1.f) : 0.f;
  if (tid < NRBF) {
    float start = __expf(-CUTR);
    float mean  = start + (float)tid * (1.f - start) / (float)(NRBF - 1);
    float bw    = 2.f / NRBF * (1.f - start);
    float beta  = 1.f / (bw * bw);
    float t     = __expf(-d) - mean;   // alpha = 5/cutoff = 1
    rb[tid]     = cc * __expf(-beta * t * t);
  }
  if (tid == 0) cut[e] = cc;
  __syncthreads();
  float acc = eeb[tid];
#pragma unroll 8
  for (int k = 0; k < NRBF; ++k) acc += rb[k] * eeW[k * Cc + tid];
  ef[(size_t)e * Cc + tid] = acc;
}

// LayerNorm over C. one block (256 thr) per node.
__global__ void k_ln(const float* __restrict__ x, const float* __restrict__ s,
                     const float* __restrict__ b, float* __restrict__ y, float eps) {
  int n = blockIdx.x, tid = threadIdx.x;
  float v = x[(size_t)n * Cc + tid];
  float sum = v, sq = v * v;
  for (int o = 16; o; o >>= 1) {
    sum += __shfl_down(sum, o, 32);
    sq  += __shfl_down(sq,  o, 32);
  }
  __shared__ float rs[8], rq[8], mb[2];
  if ((tid & 31) == 0) { rs[tid >> 5] = sum; rq[tid >> 5] = sq; }
  __syncthreads();
  if (tid == 0) {
    float ts = 0.f, tq = 0.f;
    for (int j = 0; j < 8; ++j) { ts += rs[j]; tq += rq[j]; }
    float mu = ts / Cc;
    mb[0] = mu; mb[1] = tq / Cc - mu * mu;
  }
  __syncthreads();
  y[(size_t)n * Cc + tid] = (v - mb[0]) * rsqrtf(mb[1] + eps) * s[tid] + b[tid];
}

// max_min_norm on (N,3,C). one block per node.
__global__ void k_maxmin(const float* __restrict__ nv, float* __restrict__ v, int N) {
  int n = blockIdx.x, c = threadIdx.x;
  float a = nv[((size_t)n * 3 + 0) * Cc + c];
  float b = nv[((size_t)n * 3 + 1) * Cc + c];
  float d = nv[((size_t)n * 3 + 2) * Cc + c];
  float dist = fmaxf(sqrtf(a * a + b * b + d * d), 1e-8f);
  float mx = dist, mn = dist;
  for (int o = 16; o; o >>= 1) {
    mx = fmaxf(mx, __shfl_down(mx, o, 32));
    mn = fminf(mn, __shfl_down(mn, o, 32));
  }
  __shared__ float rx[8], rn[8], mm[2];
  if ((c & 31) == 0) { rx[c >> 5] = mx; rn[c >> 5] = mn; }
  __syncthreads();
  if (c == 0) {
    float tx = rx[0], tn = rn[0];
    for (int j = 1; j < 8; ++j) { tx = fmaxf(tx, rx[j]); tn = fminf(tn, rn[j]); }
    mm[0] = tx; mm[1] = tn;
  }
  __syncthreads();
  float MX = mm[0], MN = mm[1];
  float delta = (MX == MN) ? 1.f : (MX - MN);
  float scale = (dist - MN) / (delta * dist);
  v[((size_t)n * 3 + 0) * Cc + c] = a * scale;
  v[((size_t)n * 3 + 1) * Cc + c] = b * scale;
  v[((size_t)n * 3 + 2) * Cc + c] = d * scale;
}

// attention + message. one block per edge; one wave == one head (HD==32).
__global__ void k_attn_msg(const float* __restrict__ sh, const float* __restrict__ efh,
                           const float* __restrict__ cut, const float* __restrict__ alp,
                           const int* __restrict__ snd, const int* __restrict__ rcv,
                           float* __restrict__ msg, int E) {
  int e = blockIdx.x, c = threadIdx.x;
  int s = snd[e], r = rcv[e];
  float shs = sh[(size_t)s * Cc + c];
  float eh  = efh[(size_t)e * Cc + c];
  float x   = silu_f(shs + sh[(size_t)r * Cc + c] + eh) * alp[c];
  for (int o = 16; o; o >>= 1) x += __shfl_down(x, o, 32);
  float a = __shfl(x, 0, 32) * cut[e];
  msg[(size_t)e * Cc + c] = shs * eh * a;
}

// segment_sum: msg -> ns, vmsg -> nv (atomics). one block per edge.
__global__ void k_scatter(const float* __restrict__ msg, const float* __restrict__ sca,
                          const float* __restrict__ v, const float* __restrict__ nev,
                          const int* __restrict__ snd, const int* __restrict__ rcv,
                          float* __restrict__ ns, float* __restrict__ nv, int E) {
  int e = blockIdx.x, c = threadIdx.x;
  int s = snd[e], r = rcv[e];
  atomicAdd(&ns[(size_t)r * Cc + c], msg[(size_t)e * Cc + c]);
  float s1 = sca[(size_t)e * 512 + c];
  float s2 = sca[(size_t)e * 512 + Cc + c];
#pragma unroll
  for (int d = 0; d < 3; ++d) {
    float vm = v[((size_t)s * 3 + d) * Cc + c] * s1 + s2 * nev[(size_t)e * 3 + d];
    atomicAdd(&nv[((size_t)r * 3 + d) * Cc + c], vm);
  }
}

// sum_phi = dot(cross(cv[s],u), cross(cv[r],u)) per channel.
__global__ void k_cross(const float* __restrict__ cv, const float* __restrict__ nev,
                        const int* __restrict__ snd, const int* __restrict__ rcv,
                        float* __restrict__ sphi, int E) {
  int e = blockIdx.x, c = threadIdx.x;
  int s = snd[e], r = rcv[e];
  float ux = nev[(size_t)e * 3 + 0], uy = nev[(size_t)e * 3 + 1], uz = nev[(size_t)e * 3 + 2];
  float ax = cv[((size_t)s * 3 + 0) * Cc + c];
  float ay = cv[((size_t)s * 3 + 1) * Cc + c];
  float az = cv[((size_t)s * 3 + 2) * Cc + c];
  float bx = cv[((size_t)r * 3 + 0) * Cc + c];
  float by = cv[((size_t)r * 3 + 1) * Cc + c];
  float bz = cv[((size_t)r * 3 + 2) * Cc + c];
  float cix = ay * uz - az * uy, ciy = az * ux - ax * uz, ciz = ax * uy - ay * ux;
  float cjx = by * uz - bz * uy, cjy = bz * ux - bx * uz, cjz = bx * uy - by * ux;
  sphi[(size_t)e * Cc + c] = cix * cjx + ciy * cjy + ciz * cjz;
}

__global__ void k_efupd(float* __restrict__ ef, const float* __restrict__ ft,
                        const float* __restrict__ sphi, size_t n) {
  size_t i = (size_t)blockIdx.x * blockDim.x + threadIdx.x;
  size_t stride = (size_t)gridDim.x * blockDim.x;
  for (; i < n; i += stride) ef[i] += ft[i] * sphi[i];
}

// node update from v12 (N,3,512) and sc (N,768).
__global__ void k_nodeupd(const float* __restrict__ v12, const float* __restrict__ sc,
                          float* __restrict__ ns, float* __restrict__ nv, int N) {
  int n = blockIdx.x, c = threadIdx.x;
  float v1[3], v2[3];
#pragma unroll
  for (int d = 0; d < 3; ++d) {
    v1[d] = v12[((size_t)n * 3 + d) * 512 + c];
    v2[d] = v12[((size_t)n * 3 + d) * 512 + Cc + c];
  }
  float tri = v1[0] * v2[0] + v1[1] * v2[1] + v1[2] * v2[2];
  float q   = sqrtf(v2[0] * v2[0] + v2[1] * v2[1] + v2[2] * v2[2] + 1e-8f);
  float qua = q * q * q;
  float a  = sc[(size_t)n * 768 + c];
  float b  = sc[(size_t)n * 768 + 256 + c];
  float cc = sc[(size_t)n * 768 + 512 + c];
  ns[(size_t)n * Cc + c] += (qua + tri) * a + b;
#pragma unroll
  for (int d = 0; d < 3; ++d) nv[((size_t)n * 3 + d) * Cc + c] += v1[d] * cc;
}

// final readout: out[n] = h[n,:] . r2W + r2b.  128 threads per node.
__global__ void k_out(const float* __restrict__ h, const float* __restrict__ w,
                      const float* __restrict__ b, float* __restrict__ out) {
  int n = blockIdx.x, t = threadIdx.x;
  float v = h[(size_t)n * 128 + t] * w[t];
  for (int o = 16; o; o >>= 1) v += __shfl_down(v, o, 32);
  __shared__ float r[4];
  if ((t & 31) == 0) r[t >> 5] = v;
  __syncthreads();
  if (t == 0) out[n] = r[0] + r[1] + r[2] + r[3] + b[0];
}

// ---------------------------------------------------------------------------

extern "C" void kernel_launch(void* const* d_in, const int* in_sizes, int n_in,
                              void* d_out, int out_size, void* d_ws, size_t ws_size,
                              hipStream_t stream) {
  const float* nev    = (const float*)d_in[0];
  const float* dist   = (const float*)d_in[1];
  const int*   species= (const int*)  d_in[2];
  const int*   snd    = (const int*)  d_in[3];
  const int*   rcv    = (const int*)  d_in[4];
  const float* tbl    = (const float*)d_in[5];
  const float* eeW    = (const float*)d_in[6];
  const float* eeb    = (const float*)d_in[7];
  const float* lns    = (const float*)d_in[8];
  const float* lnb    = (const float*)d_in[9];
  const float* alp    = (const float*)d_in[10];
  const float* vecW   = (const float*)d_in[11];
  const float* crossW = (const float*)d_in[12];
  const float* nodeW  = (const float*)d_in[13];
  const float* nodeb  = (const float*)d_in[14];
  const float* edgeW  = (const float*)d_in[15];
  const float* edgeb  = (const float*)d_in[16];
  const float* p1W    = (const float*)d_in[17];
  const float* p1b    = (const float*)d_in[18];
  const float* p2W    = (const float*)d_in[19];
  const float* p2b    = (const float*)d_in[20];
  const float* fW     = (const float*)d_in[21];
  const float* fb     = (const float*)d_in[22];
  const float* ons    = (const float*)d_in[23];
  const float* onb    = (const float*)d_in[24];
  const float* r1W    = (const float*)d_in[25];
  const float* r1b    = (const float*)d_in[26];
  const float* r2W    = (const float*)d_in[27];
  const float* r2b    = (const float*)d_in[28];

  const int N = in_sizes[2];
  const int E = in_sizes[1];
  const int L = 6;

  // workspace carve-up (floats)
  float* w = (float*)d_ws;
  size_t o = 0;
  float* nsB  = w + o; o += (size_t)N * Cc;
  float* nvB  = w + o; o += (size_t)N * 3 * Cc;
  float* sB   = w + o; o += (size_t)N * Cc;
  float* vB   = w + o; o += (size_t)N * 3 * Cc;
  float* shB  = w + o; o += (size_t)N * Cc;
  float* cvB  = w + o; o += (size_t)N * 3 * Cc;
  float* v12B = w + o; o += (size_t)N * 3 * 512;
  float* scB  = w + o; o += (size_t)N * 768;
  float* hB   = w + o; o += (size_t)N * 128;
  float* efB  = w + o; o += (size_t)E * Cc;
  float* efhB = w + o; o += (size_t)E * Cc;   // also reused as f-gemm tmp
  float* msgB = w + o; o += (size_t)E * Cc;   // also reused as sum_phi
  float* scaB = w + o; o += (size_t)E * 512;
  float* cutB = w + o; o += (size_t)E;
  (void)ws_size; (void)n_in; (void)out_size;

  auto gemm = [&](const float* in, const float* Wp, const float* bp, float* outp,
                  int M, int NO, int act) {
    dim3 g((M + 31) / 32, NO / 64);
    hipLaunchKernelGGL(k_gemm, g, dim3(256), 0, stream, in, Wp, bp, outp, M, NO, act);
  };

  // init
  hipLaunchKernelGGL(k_embed, dim3((N * Cc + 255) / 256), dim3(256), 0, stream,
                     species, tbl, nsB, N);
  hipLaunchKernelGGL(k_zero, dim3(2048), dim3(256), 0, stream, nvB, (size_t)N * 3 * Cc);
  hipLaunchKernelGGL(k_zero, dim3(2048), dim3(256), 0, stream, vB,  (size_t)N * 3 * Cc);
  hipLaunchKernelGGL(k_rbf_ef, dim3(E), dim3(256), 0, stream, dist, eeW, eeb, efB, cutB, E);

  for (int i = 0; i < L; ++i) {
    // s = LN(ns); v = (i==0 ? 0 : max_min_norm(nv))
    hipLaunchKernelGGL(k_ln, dim3(N), dim3(256), 0, stream,
                       nsB, lns + i * Cc, lnb + i * Cc, sB, 1e-6f);
    if (i > 0)
      hipLaunchKernelGGL(k_maxmin, dim3(N), dim3(256), 0, stream, nvB, vB, N);

    // ef_h = silu(ef @ edge_W + edge_b); s_h = s @ node_W + node_b
    gemm(efB, edgeW + (size_t)i * Cc * Cc, edgeb + i * Cc, efhB, E, Cc, 1);
    gemm(sB,  nodeW + (size_t)i * Cc * Cc, nodeb + i * Cc, shB, N, Cc, 0);

    // msg
    hipLaunchKernelGGL(k_attn_msg, dim3(E), dim3(256), 0, stream,
                       shB, efhB, cutB, alp + i * Cc, snd, rcv, msgB, E);

    // sca = silu(msg @ p1_W + p1_b); scatter ds/dv into ns/nv
    gemm(msgB, p1W + (size_t)i * Cc * 2 * Cc, p1b + i * 2 * Cc, scaB, E, 2 * Cc, 1);
    hipLaunchKernelGGL(k_scatter, dim3(E), dim3(256), 0, stream,
                       msgB, scaB, vB, nev, snd, rcv, nsB, nvB, E);

    // edge update (skipped on last layer); uses v and old ef only
    if (i < L - 1) {
      gemm(vB, crossW + (size_t)i * Cc * Cc, nullptr, cvB, 3 * N, Cc, 0);
      hipLaunchKernelGGL(k_cross, dim3(E), dim3(256), 0, stream,
                         cvB, nev, snd, rcv, msgB /*sum_phi*/, E);
      gemm(efB, fW + (size_t)i * Cc * Cc, fb + i * Cc, efhB /*tmp*/, E, Cc, 1);
      hipLaunchKernelGGL(k_efupd, dim3(4096), dim3(256), 0, stream,
                         efB, efhB, msgB, (size_t)E * Cc);
    }

    // node update
    gemm(nvB, vecW + (size_t)i * Cc * 2 * Cc, nullptr, v12B, 3 * N, 2 * Cc, 0);
    gemm(nsB, p2W + (size_t)i * Cc * 3 * Cc, p2b + i * 3 * Cc, scB, N, 3 * Cc, 0);
    hipLaunchKernelGGL(k_nodeupd, dim3(N), dim3(256), 0, stream, v12B, scB, nsB, nvB, N);
  }

  // readout
  hipLaunchKernelGGL(k_ln, dim3(N), dim3(256), 0, stream, nsB, ons, onb, sB, 1e-5f);
  gemm(sB, r1W, r1b, hB, N, 128, 1);
  hipLaunchKernelGGL(k_out, dim3(N), dim3(128), 0, stream, hB, r2W, r2b, (float*)d_out);
}
